// MultiHeadAttention_47261820125547
// MI455X (gfx1250) — matrix operand order repaired and checked by reading, not compiled, not run on hardware
//
#include <hip/hip_runtime.h>

#define D_MODEL 1024
#define NHEAD   16
#define DK      64
#define BATCH   4
#define SEQ     2048

typedef __bf16 bf16;
typedef __bf16        v16bf __attribute__((ext_vector_type(16)));
typedef __bf16        v8bf  __attribute__((ext_vector_type(8)));
typedef __bf16        v4bf  __attribute__((ext_vector_type(4)));
typedef float         v8f   __attribute__((ext_vector_type(8)));
typedef float         fx4   __attribute__((ext_vector_type(4)));
typedef unsigned int  u32x4 __attribute__((ext_vector_type(4)));

union Frag16 { u32x4 q[2]; v16bf v; };

__device__ __forceinline__ v8f v8f_zero() {
  v8f z = {0.f, 0.f, 0.f, 0.f, 0.f, 0.f, 0.f, 0.f};
  return z;
}

// A-matrix 16x32 bf16 fragment (M x K). Per ISA 7.12.2:
// lanes 0-15 / 16-31: M = row_base + (lane&15)
// lo lanes: VGPR g<4 -> K=2g..2g+1, g>=4 -> K=16+2(g-4)..; hi lanes K+=8.
// Two contiguous 16B groups at koff+hi*8 and koff+16+hi*8 (stride keeps 16B align).
__device__ __forceinline__ v16bf load_a_frag(const bf16* tile, int stride,
                                             int row_base, int koff, int lane) {
  Frag16 f;
  const int M  = row_base + (lane & 15);
  const int hi = (lane >> 4) & 1;
  const bf16* p = tile + M * stride + koff + hi * 8;
  f.q[0] = *reinterpret_cast<const u32x4*>(p);
  f.q[1] = *reinterpret_cast<const u32x4*>(p + 16);
  return f.v;
}

// B-matrix 32x16 bf16 fragment (K x N), tile stored [n][k]:
// N = col_base + (lane&15); K groups at koff+hi*8 and koff+16+hi*8 (matches A frag).
__device__ __forceinline__ v16bf load_b_frag(const bf16* tile, int stride,
                                             int col_base, int koff, int lane) {
  Frag16 f;
  const int N  = col_base + (lane & 15);
  const int hi = (lane >> 4) & 1;
  const bf16* p = tile + N * stride + koff + hi * 8;
  f.q[0] = *reinterpret_cast<const u32x4*>(p);
  f.q[1] = *reinterpret_cast<const u32x4*>(p + 16);
  return f.v;
}

#define WMMA_BF16(a, b, c) \
  __builtin_amdgcn_wmma_f32_16x16x32_bf16(false, (a), false, (b), (short)0, (c), false, false)

// ---------------------------------------------------------------- weights cvt
__global__ void cvt_bf16_kernel(const float* __restrict__ src,
                                bf16* __restrict__ dst, int n4) {
  int i = blockIdx.x * blockDim.x + threadIdx.x;
  if (i < n4) {
    fx4 f = reinterpret_cast<const fx4*>(src)[i];
    reinterpret_cast<v4bf*>(dst)[i] = __builtin_convertvector(f, v4bf);
  }
}

// ------------------------------------------------- projection GEMM (fp32 A in)
// Y[b,h,s,dk] = (X @ W^T + bias), X:[8192,1024] fp32, W:[1024,1024] bf16 row-major
// 128x64 tile, 8 waves, BK=64 (two WMMA K-steps per stage).
__global__ void __launch_bounds__(256) proj_kernel(const float* __restrict__ X,
                                                   const bf16* __restrict__ W,
                                                   const float* __restrict__ bias,
                                                   bf16* __restrict__ Y) {
  __shared__ alignas(16) bf16 As[128 * 80];
  __shared__ alignas(16) bf16 Bs[64 * 80];
  const int tid  = threadIdx.x;
  const int lane = tid & 31;
  const int wid  = tid >> 5;          // 0..7
  const int m0   = blockIdx.x * 128;
  const int n0   = blockIdx.y * 64;
  const int mw   = (wid & 3) * 32;    // wave M-subtile
  const int nw   = (wid >> 2) * 32;   // wave N-subtile
  v8f c00 = v8f_zero(), c01 = v8f_zero(), c10 = v8f_zero(), c11 = v8f_zero();

  for (int k0 = 0; k0 < D_MODEL; k0 += 64) {
    __syncthreads();
    // --- stage A: 128x64 fp32 -> bf16, batched b128 loads then b64 stores
    fx4 areg[8];
#pragma unroll
    for (int i = 0; i < 8; ++i) {
      int idx = tid + i * 256;               // 2048 float4 slots
      int r = idx >> 4, c4 = idx & 15;
      areg[i] = *reinterpret_cast<const fx4*>(&X[(m0 + r) * D_MODEL + k0 + c4 * 4]);
    }
#pragma unroll
    for (int i = 0; i < 8; ++i) {
      int idx = tid + i * 256;
      int r = idx >> 4, c4 = idx & 15;
      *reinterpret_cast<v4bf*>(&As[r * 80 + c4 * 4]) =
          __builtin_convertvector(areg[i], v4bf);
    }
    // --- stage B: 64x64 bf16, b128 load/store
    v8bf breg[2];
#pragma unroll
    for (int i = 0; i < 2; ++i) {
      int idx = tid + i * 256;               // 512 v8bf slots
      int r = idx >> 3, c8 = idx & 7;
      breg[i] = *reinterpret_cast<const v8bf*>(&W[(n0 + r) * D_MODEL + k0 + c8 * 8]);
    }
#pragma unroll
    for (int i = 0; i < 2; ++i) {
      int idx = tid + i * 256;
      int r = idx >> 3, c8 = idx & 7;
      *reinterpret_cast<v8bf*>(&Bs[r * 80 + c8 * 8]) = breg[i];
    }
    if (k0 + 64 < D_MODEL)                   // hint next A tile toward L2
      __builtin_prefetch(&X[(m0 + wid * 16) * D_MODEL + k0 + 64], 0, 1);
    __syncthreads();

#pragma unroll
    for (int kk = 0; kk < 64; kk += 32) {
      v16bf a0 = load_a_frag(As, 80, mw,      kk, lane);
      v16bf a1 = load_a_frag(As, 80, mw + 16, kk, lane);
      v16bf b0 = load_b_frag(Bs, 80, nw,      kk, lane);
      v16bf b1 = load_b_frag(Bs, 80, nw + 16, kk, lane);
      c00 = WMMA_BF16(a0, b0, c00);
      c01 = WMMA_BF16(a0, b1, c01);
      c10 = WMMA_BF16(a1, b0, c10);
      c11 = WMMA_BF16(a1, b1, c11);
    }
  }

  const int hi = (lane >> 4) & 1;
  const int lanelo = lane & 15;
#pragma unroll
  for (int f = 0; f < 4; ++f) {
    v8f acc = (f == 0) ? c00 : (f == 1) ? c01 : (f == 2) ? c10 : c11;
    int rb = mw + (f >> 1) * 16;
    int cb = nw + (f & 1) * 16;
    int n  = n0 + cb + lanelo;
    float bv = bias[n];
    int h = n >> 6, dk = n & 63;
#pragma unroll
    for (int r = 0; r < 8; ++r) {
      int m = m0 + rb + r + hi * 8;
      int b = m >> 11, s = m & (SEQ - 1);
      Y[(((long)(b * NHEAD + h) * SEQ) + s) * DK + dk] = (bf16)(acc[r] + bv);
    }
  }
}

// ------------------------------------------------------ output GEMM (bf16 A in)
__global__ void __launch_bounds__(256) oproj_kernel(const bf16* __restrict__ X,
                                                    const bf16* __restrict__ W,
                                                    const float* __restrict__ bias,
                                                    float* __restrict__ Y) {
  __shared__ alignas(16) bf16 As[128 * 80];
  __shared__ alignas(16) bf16 Bs[64 * 80];
  const int tid  = threadIdx.x;
  const int lane = tid & 31;
  const int wid  = tid >> 5;
  const int m0   = blockIdx.x * 128;
  const int n0   = blockIdx.y * 64;
  const int mw   = (wid & 3) * 32;
  const int nw   = (wid >> 2) * 32;
  v8f c00 = v8f_zero(), c01 = v8f_zero(), c10 = v8f_zero(), c11 = v8f_zero();

  for (int k0 = 0; k0 < D_MODEL; k0 += 64) {
    __syncthreads();
    v8bf areg[4];
#pragma unroll
    for (int i = 0; i < 4; ++i) {            // A: 128x64 bf16 = 1024 v8bf slots
      int idx = tid + i * 256;
      int r = idx >> 3, c8 = idx & 7;
      areg[i] = *reinterpret_cast<const v8bf*>(&X[(long)(m0 + r) * D_MODEL + k0 + c8 * 8]);
    }
#pragma unroll
    for (int i = 0; i < 4; ++i) {
      int idx = tid + i * 256;
      int r = idx >> 3, c8 = idx & 7;
      *reinterpret_cast<v8bf*>(&As[r * 80 + c8 * 8]) = areg[i];
    }
    v8bf breg[2];
#pragma unroll
    for (int i = 0; i < 2; ++i) {
      int idx = tid + i * 256;
      int r = idx >> 3, c8 = idx & 7;
      breg[i] = *reinterpret_cast<const v8bf*>(&W[(n0 + r) * D_MODEL + k0 + c8 * 8]);
    }
#pragma unroll
    for (int i = 0; i < 2; ++i) {
      int idx = tid + i * 256;
      int r = idx >> 3, c8 = idx & 7;
      *reinterpret_cast<v8bf*>(&Bs[r * 80 + c8 * 8]) = breg[i];
    }
    __syncthreads();
#pragma unroll
    for (int kk = 0; kk < 64; kk += 32) {
      v16bf a0 = load_a_frag(As, 80, mw,      kk, lane);
      v16bf a1 = load_a_frag(As, 80, mw + 16, kk, lane);
      v16bf b0 = load_b_frag(Bs, 80, nw,      kk, lane);
      v16bf b1 = load_b_frag(Bs, 80, nw + 16, kk, lane);
      c00 = WMMA_BF16(a0, b0, c00);
      c01 = WMMA_BF16(a0, b1, c01);
      c10 = WMMA_BF16(a1, b0, c10);
      c11 = WMMA_BF16(a1, b1, c11);
    }
  }

  const int hi = (lane >> 4) & 1;
  const int lanelo = lane & 15;
#pragma unroll
  for (int f = 0; f < 4; ++f) {
    v8f acc = (f == 0) ? c00 : (f == 1) ? c01 : (f == 2) ? c10 : c11;
    int rb = mw + (f >> 1) * 16;
    int cb = nw + (f & 1) * 16;
    int n  = n0 + cb + lanelo;
    float bv = bias[n];
#pragma unroll
    for (int r = 0; r < 8; ++r) {
      int m = m0 + rb + r + hi * 8;
      Y[(long)m * D_MODEL + n] = acc[r] + bv;
    }
  }
}

// --------------------------------------------------------- flash attention
// One workgroup (4 waves) per (b*h, 64-row q tile). Q/K/V in [B,H,S,DK] bf16.
__global__ void __launch_bounds__(128) attn_kernel(const bf16* __restrict__ Qp,
                                                   const bf16* __restrict__ Kp,
                                                   const bf16* __restrict__ Vp,
                                                   bf16* __restrict__ Xa) {
  __shared__ alignas(16) bf16 Qs[64 * 80];
  __shared__ alignas(16) bf16 Ks[64 * 80];
  __shared__ alignas(16) bf16 Vt[64 * 80];   // V transposed: [d][key]
  __shared__ alignas(16) bf16 Ps[64 * 80];   // probabilities, per-wave rows
  const int tid    = threadIdx.x;
  const int lane   = tid & 31;
  const int wid    = tid >> 5;   // 0..3, wave owns rows wid*16..+15
  const int lanelo = lane & 15;
  const int hi     = (lane >> 4) & 1;
  const int qt     = blockIdx.x;                 // 0..31
  const int bh     = blockIdx.y;                 // 0..63
  const int q0     = qt * 64;
  const long baseP = (long)bh * SEQ * DK;

  {                                              // stage Q 64x64, b128 moves
    v8bf qreg[4];
#pragma unroll
    for (int i = 0; i < 4; ++i) {
      int idx = tid + i * 128;                   // 512 v8bf slots
      int r = idx >> 3, c8 = idx & 7;
      qreg[i] = *reinterpret_cast<const v8bf*>(&Qp[baseP + (long)(q0 + r) * DK + c8 * 8]);
    }
#pragma unroll
    for (int i = 0; i < 4; ++i) {
      int idx = tid + i * 128;
      int r = idx >> 3, c8 = idx & 7;
      *reinterpret_cast<v8bf*>(&Qs[r * 80 + c8 * 8]) = qreg[i];
    }
  }
  __syncthreads();
  v16bf aq0 = load_a_frag(Qs, 80, wid * 16, 0,  lane);
  v16bf aq1 = load_a_frag(Qs, 80, wid * 16, 32, lane);

  v8f oacc[4];
  float mrow[8], lrow[8];
#pragma unroll
  for (int t = 0; t < 4; ++t) oacc[t] = v8f_zero();
#pragma unroll
  for (int r = 0; r < 8; ++r) { mrow[r] = -__builtin_inff(); lrow[r] = 0.f; }
  const float LOG2E = 1.44269504088896340736f;
  const float SCALE = 0.125f;                    // 1/sqrt(64)

  const int ntiles = qt + 1;                     // causal: skip tiles past diag
  for (int jt = 0; jt < ntiles; ++jt) {
    const int j0 = jt * 64;
    __syncthreads();
    {                                            // stage K [key][d], V^T [d][key]
      v8bf kreg[4], vreg[4];
#pragma unroll
      for (int i = 0; i < 4; ++i) {
        int idx = tid + i * 128;
        int r = idx >> 3, c8 = idx & 7;
        kreg[i] = *reinterpret_cast<const v8bf*>(&Kp[baseP + (long)(j0 + r) * DK + c8 * 8]);
        vreg[i] = *reinterpret_cast<const v8bf*>(&Vp[baseP + (long)(j0 + r) * DK + c8 * 8]);
      }
#pragma unroll
      for (int i = 0; i < 4; ++i) {
        int idx = tid + i * 128;
        int r = idx >> 3, c8 = idx & 7;
        *reinterpret_cast<v8bf*>(&Ks[r * 80 + c8 * 8]) = kreg[i];
#pragma unroll
        for (int e = 0; e < 8; ++e)
          Vt[(c8 * 8 + e) * 80 + r] = vreg[i][e];
      }
    }
    __syncthreads();

    // scores: S = Q @ K^T  (B frag: [n=key][k=d] == Ks layout)
    float sc[4][8];
#pragma unroll
    for (int t = 0; t < 4; ++t) {
      v16bf bk0 = load_b_frag(Ks, 80, t * 16, 0,  lane);
      v16bf bk1 = load_b_frag(Ks, 80, t * 16, 32, lane);
      v8f s = v8f_zero();
      s = WMMA_BF16(aq0, bk0, s);
      s = WMMA_BF16(aq1, bk1, s);
      const int kidx = j0 + t * 16 + lanelo;
#pragma unroll
      for (int r = 0; r < 8; ++r) {
        int qidx = q0 + wid * 16 + r + hi * 8;
        sc[t][r] = (kidx > qidx) ? -__builtin_inff() : s[r] * SCALE;
      }
    }

    // online softmax; rows live in 16-lane halves (xor 1/2/4/8 stays in half)
#pragma unroll
    for (int r = 0; r < 8; ++r) {
      float mx = fmaxf(fmaxf(sc[0][r], sc[1][r]), fmaxf(sc[2][r], sc[3][r]));
      mx = fmaxf(mx, __shfl_xor(mx, 1, 32));
      mx = fmaxf(mx, __shfl_xor(mx, 2, 32));
      mx = fmaxf(mx, __shfl_xor(mx, 4, 32));
      mx = fmaxf(mx, __shfl_xor(mx, 8, 32));
      float mnew  = fmaxf(mrow[r], mx);
      float alpha = exp2f((mrow[r] - mnew) * LOG2E);
      int row = wid * 16 + r + hi * 8;
      float rsum = 0.f;
#pragma unroll
      for (int t = 0; t < 4; ++t) {
        float p = exp2f((sc[t][r] - mnew) * LOG2E);
        rsum += p;
        Ps[row * 80 + t * 16 + lanelo] = (bf16)p;
      }
      rsum += __shfl_xor(rsum, 1, 32);
      rsum += __shfl_xor(rsum, 2, 32);
      rsum += __shfl_xor(rsum, 4, 32);
      rsum += __shfl_xor(rsum, 8, 32);
      lrow[r] = lrow[r] * alpha + rsum;
      mrow[r] = mnew;
#pragma unroll
      for (int t = 0; t < 4; ++t) oacc[t][r] *= alpha;
    }

    // wave-private P rows: DS ops are in-order per wave; wait to be explicit
    asm volatile("s_wait_dscnt 0" ::: "memory");
    v16bf ap0 = load_a_frag(Ps, 80, wid * 16, 0,  lane);
    v16bf ap1 = load_a_frag(Ps, 80, wid * 16, 32, lane);
#pragma unroll
    for (int t = 0; t < 4; ++t) {                // O += P @ V
      v16bf bv0 = load_b_frag(Vt, 80, t * 16, 0,  lane);
      v16bf bv1 = load_b_frag(Vt, 80, t * 16, 32, lane);
      oacc[t] = WMMA_BF16(ap0, bv0, oacc[t]);
      oacc[t] = WMMA_BF16(ap1, bv1, oacc[t]);
    }
  }

  const int b = bh >> 4, h = bh & 15;
#pragma unroll
  for (int t = 0; t < 4; ++t) {
#pragma unroll
    for (int r = 0; r < 8; ++r) {
      int row = q0 + wid * 16 + r + hi * 8;
      int col = h * 64 + t * 16 + lanelo;
      Xa[(long)(b * SEQ + row) * D_MODEL + col] = (bf16)(oacc[t][r] / lrow[r]);
    }
  }
}

// ------------------------------------------------------------------- launcher
extern "C" void kernel_launch(void* const* d_in, const int* in_sizes, int n_in,
                              void* d_out, int out_size, void* d_ws, size_t ws_size,
                              hipStream_t stream) {
  (void)in_sizes; (void)n_in; (void)out_size; (void)ws_size;
  const float* q   = (const float*)d_in[0];
  const float* k   = (const float*)d_in[1];
  const float* v   = (const float*)d_in[2];
  const float* w_q = (const float*)d_in[3];
  const float* b_q = (const float*)d_in[4];
  const float* w_k = (const float*)d_in[5];
  const float* b_k = (const float*)d_in[6];
  const float* w_v = (const float*)d_in[7];
  const float* b_v = (const float*)d_in[8];
  const float* w_o = (const float*)d_in[9];
  const float* b_o = (const float*)d_in[10];
  float* out = (float*)d_out;

  char* ws = (char*)d_ws;
  const size_t WBYTES = (size_t)D_MODEL * D_MODEL * sizeof(bf16);      // 2 MiB
  const size_t TBYTES = (size_t)BATCH * SEQ * D_MODEL * sizeof(bf16);  // 16 MiB
  bf16* Wq = (bf16*)(ws + 0 * WBYTES);
  bf16* Wk = (bf16*)(ws + 1 * WBYTES);
  bf16* Wv = (bf16*)(ws + 2 * WBYTES);
  bf16* Wo = (bf16*)(ws + 3 * WBYTES);
  bf16* Qp = (bf16*)(ws + 4 * WBYTES + 0 * TBYTES);
  bf16* Kp = (bf16*)(ws + 4 * WBYTES + 1 * TBYTES);
  bf16* Vp = (bf16*)(ws + 4 * WBYTES + 2 * TBYTES);
  bf16* Xa = (bf16*)(ws + 4 * WBYTES + 3 * TBYTES);

  const int wn4 = D_MODEL * D_MODEL / 4;
  cvt_bf16_kernel<<<(wn4 + 255) / 256, 256, 0, stream>>>(w_q, Wq, wn4);
  cvt_bf16_kernel<<<(wn4 + 255) / 256, 256, 0, stream>>>(w_k, Wk, wn4);
  cvt_bf16_kernel<<<(wn4 + 255) / 256, 256, 0, stream>>>(w_v, Wv, wn4);
  cvt_bf16_kernel<<<(wn4 + 255) / 256, 256, 0, stream>>>(w_o, Wo, wn4);

  dim3 gg(BATCH * SEQ / 128, D_MODEL / 64), gb(256);
  proj_kernel<<<gg, gb, 0, stream>>>(q, Wq, b_q, Qp);
  proj_kernel<<<gg, gb, 0, stream>>>(k, Wk, b_k, Kp);
  proj_kernel<<<gg, gb, 0, stream>>>(v, Wv, b_v, Vp);

  attn_kernel<<<dim3(SEQ / 64, BATCH * NHEAD), 128, 0, stream>>>(Qp, Kp, Vp, Xa);

  oproj_kernel<<<gg, gb, 0, stream>>>(Xa, Wo, b_o, out);
}
